// SeqDetect_32074815767079
// MI455X (gfx1250) — matrix operand order repaired and checked by reading, not compiled, not run on hardware
//
#include <hip/hip_runtime.h>
#include <hip/hip_bf16.h>

// Sizes from the reference
#define BATCH     16384
#define IN_SZ     512
#define HID       128
#define OUTSZ     64
#define NSTEPS    10
#define FOURH     512      // 4*HID

#define ROWS_PER_WG   128   // 8 waves * 16 rows
#define THREADS       256

typedef __attribute__((ext_vector_type(4)))  float     v4f;
typedef __attribute__((ext_vector_type(8)))  float     v8f;
typedef __attribute__((ext_vector_type(8)))  _Float16  v8h;
typedef __attribute__((ext_vector_type(16))) _Float16  v16h;

#define SHUF16(lo,hi) __builtin_shufflevector((lo),(hi),0,1,2,3,4,5,6,7,8,9,10,11,12,13,14,15)
#define WMMA_F16(a,b,c) __builtin_amdgcn_wmma_f32_16x16x32_f16(false,(a),false,(b),(short)0,(c),false,false)

// ---- LDS layout (bytes) -------------------------------------------------
#define OFF_WIH   0                           // [512][64]  f16 : 65536
#define OFF_WHH   (OFF_WIH  + FOURH*OUTSZ*2)  // [512][128] f16 : 131072
#define OFF_WOUT  (OFF_WHH  + FOURH*HID*2)    // [64][128]  f16 : 16384
#define OFF_HX    (OFF_WOUT + OUTSZ*HID*2)    // [128][128] f16 : 32768
#define OFF_O     (OFF_HX   + ROWS_PER_WG*HID*2)   // [128][64] f16 : 16384
#define OFF_BIAS  (OFF_O    + ROWS_PER_WG*OUTSZ*2) // [512] f32    : 2048
#define OFF_BOUT  (OFF_BIAS + FOURH*4)             // [64]  f32    : 256
#define LDS_BYTES (OFF_BOUT + OUTSZ*4)             // 264448 total

// ---- helpers ------------------------------------------------------------
__device__ __forceinline__ v8h cvt8f(const float* p) {
    v4f a = *(const v4f*)p;
    v4f b = *(const v4f*)(p + 4);
    v8h r;
#pragma unroll
    for (int i = 0; i < 4; ++i) { r[i] = (_Float16)a[i]; r[i + 4] = (_Float16)b[i]; }
    return r;
}
// A fragment (16x32, M x K) from f32 global: lane row = M, chunks at K +0..7 and +16..23
__device__ __forceinline__ v16h load_a_f32(const float* p) { return SHUF16(cvt8f(p), cvt8f(p + 16)); }
// B fragment (32x16, K x N) from f32 global: lane col = N, chunks at K +0..7 and +16..23
__device__ __forceinline__ v16h load_b_f32(const float* p) { return SHUF16(cvt8f(p), cvt8f(p + 16)); }
// Same from f16 LDS
__device__ __forceinline__ v16h load_a_lds(const _Float16* p) {
    v8h lo = *(const v8h*)p; v8h hi = *(const v8h*)(p + 16); return SHUF16(lo, hi);
}
__device__ __forceinline__ v16h load_b_lds(const _Float16* p) {
    v8h lo = *(const v8h*)p; v8h hi = *(const v8h*)(p + 16); return SHUF16(lo, hi);
}
__device__ __forceinline__ float sigm_f(float x) { return 1.0f / (1.0f + __expf(-x)); }
__device__ __forceinline__ float tanh_f(float x) {
    float e = __expf(-2.0f * x);
    return (1.0f - e) / (1.0f + e);
}

// ---- kernel -------------------------------------------------------------
__global__ void __launch_bounds__(THREADS, 1)
lstm_seq_wmma(const float* __restrict__ x,
              const float* __restrict__ W_ih,  const float* __restrict__ W_hh,
              const float* __restrict__ b_ih,  const float* __restrict__ b_hh,
              const float* __restrict__ W_init,const float* __restrict__ b_init,
              const float* __restrict__ W_out, const float* __restrict__ b_out,
              float* __restrict__ out)
{
    extern __shared__ char smem[];
    _Float16* sWih  = (_Float16*)(smem + OFF_WIH);
    _Float16* sWhh  = (_Float16*)(smem + OFF_WHH);
    _Float16* sWout = (_Float16*)(smem + OFF_WOUT);
    _Float16* sHx   = (_Float16*)(smem + OFF_HX);
    _Float16* sO    = (_Float16*)(smem + OFF_O);
    float*    sBias = (float*)   (smem + OFF_BIAS);
    float*    sBout = (float*)   (smem + OFF_BOUT);

    const int tid  = threadIdx.x;
    const int lane = tid & 31;
    const int wave = tid >> 5;          // 0..7, each wave owns 16 batch rows
    const int lm   = lane & 15;         // sub-tile row / col index
    const int lh   = lane >> 4;         // 0 or 1 (K half selector)
    const int koffA = lh * 8;           // A-fragment K sub-offset
    const int koffB = lh * 8;           // B-fragment K sub-offset
    const int wrow  = wave * 16;        // wave's row base inside the WG tile

    // ---- cooperative setup: weights f32 -> f16 LDS, fused bias, o0 = 0 --
    for (int i = tid; i < FOURH * OUTSZ; i += THREADS) sWih[i]  = (_Float16)W_ih[i];
    for (int i = tid; i < FOURH * HID;   i += THREADS) sWhh[i]  = (_Float16)W_hh[i];
    for (int i = tid; i < OUTSZ * HID;   i += THREADS) sWout[i] = (_Float16)W_out[i];
    for (int i = tid; i < FOURH;         i += THREADS) sBias[i] = b_ih[i] + b_hh[i];
    for (int i = tid; i < OUTSZ;         i += THREADS) sBout[i] = b_out[i];
    for (int i = tid; i < ROWS_PER_WG * OUTSZ; i += THREADS) sO[i] = (_Float16)0.0f;
    __syncthreads();

    const int rowBase = blockIdx.x * ROWS_PER_WG + wrow;   // global batch row base

    // cx held in registers in WMMA accumulator layout: 8 tiles of [16 x 16]
    v8f cacc[8];
#pragma unroll
    for (int ct = 0; ct < 8; ++ct) cacc[ct] = (v8f)(0.0f);

    // ---- init GEMM: hx = x @ W_init.T + b_init, K = 512 ----------------
#pragma unroll 1
    for (int kt = 0; kt < IN_SZ / 32; ++kt) {
        const float* px = x + (size_t)(rowBase + lm) * IN_SZ + kt * 32 + koffA;
        v16h a = load_a_f32(px);
#pragma unroll
        for (int ct = 0; ct < 8; ++ct) {
            const float* pw = W_init + (size_t)(ct * 16 + lm) * IN_SZ + kt * 32 + koffB;
            v16h b = load_b_f32(pw);
            cacc[ct] = WMMA_F16(a, b, cacc[ct]);
        }
    }
#pragma unroll
    for (int ct = 0; ct < 8; ++ct) {
        float bi = b_init[ct * 16 + lm];
#pragma unroll
        for (int r = 0; r < 8; ++r) {
            cacc[ct][r] += bi;
            sHx[(wrow + r + lh * 8) * HID + ct * 16 + lm] = (_Float16)cacc[ct][r];
        }
    }
    __syncthreads();

    // ---- 10 recurrent steps --------------------------------------------
#pragma unroll 1
    for (int t = 0; t < NSTEPS; ++t) {
        // A fragments for gates GEMM: K = [o(0..63) | hx(0..127)] = 6 K-tiles
        v16h afr[6];
        afr[0] = load_a_lds(sO + (wrow + lm) * OUTSZ + 0  + koffA);
        afr[1] = load_a_lds(sO + (wrow + lm) * OUTSZ + 32 + koffA);
#pragma unroll
        for (int k = 0; k < 4; ++k)
            afr[2 + k] = load_a_lds(sHx + (wrow + lm) * HID + k * 32 + koffA);

#pragma unroll
        for (int ct = 0; ct < 8; ++ct) {
            v8f g4[4];
#pragma unroll
            for (int gi = 0; gi < 4; ++gi) {        // gate order: i, f, g, o
                const int J = gi * HID + ct * 16;   // gates column base
                v8f acc = (v8f)(0.0f);
#pragma unroll
                for (int kk = 0; kk < 6; ++kk) {
                    const _Float16* pb = (kk < 2)
                        ? sWih + (J + lm) * OUTSZ + kk * 32 + koffB
                        : sWhh + (J + lm) * HID + (kk - 2) * 32 + koffB;
                    acc = WMMA_F16(afr[kk], load_b_lds(pb), acc);
                }
                float bb = sBias[J + lm];
#pragma unroll
                for (int r = 0; r < 8; ++r) acc[r] += bb;
                g4[gi] = acc;
            }
            // elementwise LSTM cell update, all in accumulator layout
#pragma unroll
            for (int r = 0; r < 8; ++r) {
                float iv = sigm_f(g4[0][r]);
                float fv = sigm_f(g4[1][r]);
                float gv = tanh_f(g4[2][r]);
                float ov = sigm_f(g4[3][r]);
                float c  = fv * cacc[ct][r] + iv * gv;
                cacc[ct][r] = c;
                float h  = ov * tanh_f(c);
                sHx[(wrow + r + lh * 8) * HID + ct * 16 + lm] = (_Float16)h;
            }
        }
        __syncthreads();   // hx complete for this step

        // ---- output GEMM: o = hx @ W_out.T + b_out, K = 128 ------------
#pragma unroll
        for (int nt = 0; nt < 4; ++nt) {
            v8f acc = (v8f)(0.0f);
#pragma unroll
            for (int kk = 0; kk < 4; ++kk) {
                v16h a = load_a_lds(sHx + (wrow + lm) * HID + kk * 32 + koffA);
                v16h b = load_b_lds(sWout + (nt * 16 + lm) * HID + kk * 32 + koffB);
                acc = WMMA_F16(a, b, acc);
            }
            float bb = sBout[nt * 16 + lm];
#pragma unroll
            for (int r = 0; r < 8; ++r) {
                float v = acc[r] + bb;
                int rowG = rowBase + r + lh * 8;
                out[(size_t)rowG * (NSTEPS * OUTSZ) + t * OUTSZ + nt * 16 + lm] = v;
                sO[(wrow + r + lh * 8) * OUTSZ + nt * 16 + lm] = (_Float16)v;
            }
        }
        __syncthreads();   // o ready for next step's A fragments
    }
}

extern "C" void kernel_launch(void* const* d_in, const int* in_sizes, int n_in,
                              void* d_out, int out_size, void* d_ws, size_t ws_size,
                              hipStream_t stream) {
    const float* x      = (const float*)d_in[0];
    const float* W_ih   = (const float*)d_in[1];
    const float* W_hh   = (const float*)d_in[2];
    const float* b_ih   = (const float*)d_in[3];
    const float* b_hh   = (const float*)d_in[4];
    const float* W_init = (const float*)d_in[5];
    const float* b_init = (const float*)d_in[6];
    const float* W_out  = (const float*)d_in[7];
    const float* b_out  = (const float*)d_in[8];
    float* out = (float*)d_out;

    dim3 grid(BATCH / ROWS_PER_WG);   // 128 workgroups
    dim3 block(THREADS);              // 8 waves of 32
    lstm_seq_wmma<<<grid, block, LDS_BYTES, stream>>>(
        x, W_ih, W_hh, b_ih, b_hh, W_init, b_init, W_out, b_out, out);
}
